// FutureCameraHead_78434692760090
// MI455X (gfx1250) — matrix operand order repaired and checked by reading, not compiled, not run on hardware
//
#include <hip/hip_runtime.h>
#include <hip/hip_bf16.h>
#include <stdint.h>

typedef __attribute__((ext_vector_type(16))) __bf16 v16bf;
typedef __attribute__((ext_vector_type(8)))  float  v8f;

#define DDIM 512
#define HW   1369
#define BNUM 96
#define ROWS (BNUM * HW)   /* 131424, divisible by 32 */
#define NEWBN 192

static __device__ __forceinline__ float bf2f(unsigned short h) {
  return __uint_as_float(((unsigned int)h) << 16);
}
static __device__ __forceinline__ unsigned short f2bf(float f) {
  unsigned int u = __float_as_uint(f);
  unsigned int r = u + 0x7FFFu + ((u >> 16) & 1u);          // round-to-nearest-even
  if (((u >> 23) & 0xFFu) == 0xFFu) r = u;                  // Inf/NaN passthrough
  return (unsigned short)(r >> 16);
}
static __device__ __forceinline__ unsigned int pack2(float a, float b) {
  return (unsigned int)f2bf(a) | ((unsigned int)f2bf(b) << 16);
}

// ---------------- f32 -> bf16 bulk convert (feat) ----------------
__global__ void k_cvt_bf16(const float* __restrict__ in,
                           unsigned short* __restrict__ out, int n4) {
  int i = blockIdx.x * blockDim.x + threadIdx.x;
  if (i >= n4) return;
  float4 f = ((const float4*)in)[i];
  uint2 o;
  o.x = pack2(f.x, f.y);
  o.y = pack2(f.z, f.w);
  ((uint2*)out)[i] = o;
}

// ---- W_res [6][512 in][512 out] f32 -> Wt [6][512 out][512 in] bf16 ----
__global__ void k_wt_bf16(const float* __restrict__ W,
                          unsigned short* __restrict__ Wt, int total) {
  int i = blockIdx.x * blockDim.x + threadIdx.x;
  if (i >= total) return;
  int l = i >> 18;
  int r = i & 262143;
  int n = r >> 9;      // out col
  int k = r & 511;     // in  (K)
  Wt[(size_t)l * 262144 + (size_t)n * 512 + k] =
      f2bf(W[(size_t)l * 262144 + (size_t)k * 512 + n]);
}

// ------------- big GEMM: Y = act(X[rows,512] @ W + b), bf16 WMMA -------------
// mode 0: Y = relu(XW+b)        mode 1: Y = bf16(f32(Y) + relu(XW+b))  (residual)
union FragU { uint4 q[2]; v16bf v; };

__global__ __launch_bounds__(256) void k_gemm_wmma(
    const unsigned short* __restrict__ X, const unsigned short* __restrict__ Wt,
    const float* __restrict__ bias, unsigned short* __restrict__ Y, int mode) {
  __shared__ unsigned short As[32 * 520];   // 32 rows x 512 K, padded stride 520
  const int tid  = threadIdx.x;
  const int w    = tid >> 5;      // wave 0..7 -> 64-col slice
  const int lane = tid & 31;
  const int lm   = lane & 15;
  const int half = lane >> 4;
  const int row0 = blockIdx.x * 32;

  // stage 32 x-rows into LDS (16B vector copies)
  for (int c = tid; c < 2048; c += 256) {
    int r  = c >> 6;
    int cc = c & 63;
    uint4 v = *(const uint4*)(X + (size_t)(row0 + r) * DDIM + cc * 8);
    *(uint4*)(&As[r * 520 + cc * 8]) = v;
  }
  __syncthreads();

  v8f acc[4][2] = {};   // 4 N-tiles x 2 M-tiles, f32 accum

  for (int k0 = 0; k0 < DDIM; k0 += 32) {
    FragU fa[4], fb[2];
#pragma unroll
    for (int i = 0; i < 4; ++i) {           // A = Wt tile rows (output cols)
      const unsigned short* pg =
          Wt + (size_t)(w * 64 + i * 16 + lm) * DDIM + (k0 + 8 * half);
      fa[i].q[0] = *(const uint4*)pg;
      fa[i].q[1] = *(const uint4*)(pg + 16);
    }
#pragma unroll
    for (int t = 0; t < 2; ++t) {           // B = x tile (cols = x rows)
      const unsigned short* ps = &As[(t * 16 + lm) * 520 + k0 + 8 * half];
      fb[t].q[0] = *(const uint4*)ps;
      fb[t].q[1] = *(const uint4*)(ps + 16);
    }
#pragma unroll
    for (int i = 0; i < 4; ++i)
#pragma unroll
      for (int t = 0; t < 2; ++t)
        acc[i][t] = __builtin_amdgcn_wmma_f32_16x16x32_bf16(
            false, fa[i].v, false, fb[t].v, (short)0, acc[i][t], false, false);
  }

  // epilogue: lane holds y[row m][nb..nb+7]
#pragma unroll
  for (int t = 0; t < 2; ++t)
#pragma unroll
    for (int i = 0; i < 4; ++i) {
      int m  = row0 + t * 16 + lm;
      int nb = w * 64 + i * 16 + 8 * half;
      float4 b0 = *(const float4*)(bias + nb);
      float4 b1 = *(const float4*)(bias + nb + 4);
      float v[8];
      v[0] = acc[i][t][0] + b0.x; v[1] = acc[i][t][1] + b0.y;
      v[2] = acc[i][t][2] + b0.z; v[3] = acc[i][t][3] + b0.w;
      v[4] = acc[i][t][4] + b1.x; v[5] = acc[i][t][5] + b1.y;
      v[6] = acc[i][t][6] + b1.z; v[7] = acc[i][t][7] + b1.w;
#pragma unroll
      for (int j = 0; j < 8; ++j) v[j] = fmaxf(v[j], 0.f);
      unsigned short* dst = Y + (size_t)m * DDIM + nb;
      if (mode == 1) {
        uint4 old = *(uint4*)dst;
        v[0] += bf2f((unsigned short)(old.x & 0xffff));
        v[1] += bf2f((unsigned short)(old.x >> 16));
        v[2] += bf2f((unsigned short)(old.y & 0xffff));
        v[3] += bf2f((unsigned short)(old.y >> 16));
        v[4] += bf2f((unsigned short)(old.z & 0xffff));
        v[5] += bf2f((unsigned short)(old.z >> 16));
        v[6] += bf2f((unsigned short)(old.w & 0xffff));
        v[7] += bf2f((unsigned short)(old.w >> 16));
      }
      uint4 o;
      o.x = pack2(v[0], v[1]); o.y = pack2(v[2], v[3]);
      o.z = pack2(v[4], v[5]); o.w = pack2(v[6], v[7]);
      *(uint4*)dst = o;
    }
}

// ---------------- mean over hw: pooled[bn][d] ----------------
__global__ void k_pool(const unsigned short* __restrict__ xb,
                       float* __restrict__ pooled) {
  int bn = blockIdx.x;
  int d  = threadIdx.x;   // 512 threads
  const unsigned short* p = xb + (size_t)bn * HW * DDIM + d;
  float s = 0.f;
  for (int i = 0; i < HW; ++i) s += bf2f(p[(size_t)i * DDIM]);
  pooled[bn * DDIM + d] = s * (1.0f / (float)HW);
}

// ---------------- mean over N=3 frames ----------------
__global__ void k_gctx(const float* __restrict__ pooled, float* __restrict__ g) {
  int i = blockIdx.x * blockDim.x + threadIdx.x;
  if (i >= 32 * DDIM) return;
  int b = i >> 9, d = i & 511;
  g[i] = (pooled[(b * 3 + 0) * DDIM + d] + pooled[(b * 3 + 1) * DDIM + d] +
          pooled[(b * 3 + 2) * DDIM + d]) * (1.f / 3.f);
}

// ------------- small exact-fp32 GEMM: C = act(A[M,K] @ W[K,N] + b) -------------
__global__ void k_sgemm(const float* __restrict__ A, const float* __restrict__ W,
                        const float* __restrict__ bias, float* __restrict__ C,
                        int M, int N, int K, int relu) {
  int m = blockIdx.x;
  if (m >= M) return;
  const float* a = A + (size_t)m * K;
  for (int n = threadIdx.x; n < N; n += blockDim.x) {
    float s = bias[n];
    for (int k = 0; k < K; ++k) s = fmaf(a[k], W[(size_t)k * N + n], s);
    if (relu) s = fmaxf(s, 0.f);
    C[(size_t)m * N + n] = s;
  }
}

// ---------- special Procrustes (SVD orthogonalize) + pose assembly ----------
__global__ void k_pose(const float* __restrict__ tvec, const float* __restrict__ rvec,
                       float* __restrict__ pose, int n) {
  int i = blockIdx.x * blockDim.x + threadIdx.x;
  if (i >= n) return;
  float A[3][3];
#pragma unroll
  for (int r = 0; r < 3; ++r) {
    float x = rvec[i * 9 + r * 3 + 0];
    float y = rvec[i * 9 + r * 3 + 1];
    float z = rvec[i * 9 + r * 3 + 2];
    float inv = 1.f / fmaxf(sqrtf(x * x + y * y + z * z), 1e-12f);
    A[r][0] = x * inv; A[r][1] = y * inv; A[r][2] = z * inv;
  }
  float H[3][3];
#pragma unroll
  for (int a = 0; a < 3; ++a)
#pragma unroll
    for (int b = 0; b < 3; ++b)
      H[a][b] = A[0][a] * A[0][b] + A[1][a] * A[1][b] + A[2][a] * A[2][b];
  float V[3][3] = {{1, 0, 0}, {0, 1, 0}, {0, 0, 1}};
  for (int sweep = 0; sweep < 8; ++sweep)
    for (int pi = 0; pi < 3; ++pi) {
      int p = (pi == 2) ? 1 : 0;
      int q = (pi == 0) ? 1 : 2;
      float apq = H[p][q];
      if (fabsf(apq) < 1e-18f) continue;
      float tau = (H[q][q] - H[p][p]) / (2.f * apq);
      float tt = (tau >= 0.f ? 1.f : -1.f) / (fabsf(tau) + sqrtf(1.f + tau * tau));
      float c = rsqrtf(1.f + tt * tt);
      float s = tt * c;
      for (int k = 0; k < 3; ++k) {
        float hpk = H[p][k], hqk = H[q][k];
        H[p][k] = c * hpk - s * hqk;
        H[q][k] = s * hpk + c * hqk;
      }
      for (int k = 0; k < 3; ++k) {
        float hkp = H[k][p], hkq = H[k][q];
        H[k][p] = c * hkp - s * hkq;
        H[k][q] = s * hkp + c * hkq;
        float vkp = V[k][p], vkq = V[k][q];
        V[k][p] = c * vkp - s * vkq;
        V[k][q] = s * vkp + c * vkq;
      }
    }
  float lam[3] = {H[0][0], H[1][1], H[2][2]};
  for (int a = 0; a < 2; ++a)
    for (int b = a + 1; b < 3; ++b)
      if (lam[b] > lam[a]) {
        float tl = lam[a]; lam[a] = lam[b]; lam[b] = tl;
        for (int r = 0; r < 3; ++r) { float tv = V[r][a]; V[r][a] = V[r][b]; V[r][b] = tv; }
      }
  float U[3][3];
#pragma unroll
  for (int m = 0; m < 3; ++m) {
    float inv = 1.f / fmaxf(sqrtf(fmaxf(lam[m], 0.f)), 1e-12f);
#pragma unroll
    for (int r = 0; r < 3; ++r)
      U[r][m] = (A[r][0] * V[0][m] + A[r][1] * V[1][m] + A[r][2] * V[2][m]) * inv;
  }
  float det = A[0][0] * (A[1][1] * A[2][2] - A[1][2] * A[2][1])
            - A[0][1] * (A[1][0] * A[2][2] - A[1][2] * A[2][0])
            + A[0][2] * (A[1][0] * A[2][1] - A[1][1] * A[2][0]);
  float dsgn = (det >= 0.f) ? 1.f : -1.f;
  float* o = pose + (size_t)i * 16;
#pragma unroll
  for (int r = 0; r < 3; ++r) {
#pragma unroll
    for (int c = 0; c < 3; ++c)
      o[r * 4 + c] = U[r][0] * V[c][0] + U[r][1] * V[c][1] + dsgn * U[r][2] * V[c][2];
    o[r * 4 + 3] = tvec[i * 3 + r];
  }
  o[12] = 0.f; o[13] = 0.f; o[14] = 0.f; o[15] = 1.f;
}

extern "C" void kernel_launch(void* const* d_in, const int* in_sizes, int n_in,
                              void* d_out, int out_size, void* d_ws, size_t ws_size,
                              hipStream_t stream) {
  (void)in_sizes; (void)n_in; (void)out_size; (void)ws_size;
  const float* feat   = (const float*)d_in[0];
  const float* W_res  = (const float*)d_in[5];
  const float* b_res  = (const float*)d_in[6];
  const float* W_cur1 = (const float*)d_in[7];  const float* b_cur1 = (const float*)d_in[8];
  const float* W_cur2 = (const float*)d_in[9];  const float* b_cur2 = (const float*)d_in[10];
  const float* W_tp1  = (const float*)d_in[11]; const float* b_tp1  = (const float*)d_in[12];
  const float* W_tp2  = (const float*)d_in[13]; const float* b_tp2  = (const float*)d_in[14];
  const float* W_fut1 = (const float*)d_in[15]; const float* b_fut1 = (const float*)d_in[16];
  const float* W_fut2 = (const float*)d_in[17]; const float* b_fut2 = (const float*)d_in[18];
  const float* W_m1   = (const float*)d_in[19]; const float* b_m1   = (const float*)d_in[20];
  const float* W_m2   = (const float*)d_in[21]; const float* b_m2   = (const float*)d_in[22];
  const float* W_t    = (const float*)d_in[23]; const float* b_t    = (const float*)d_in[24];
  const float* W_r    = (const float*)d_in[25]; const float* b_r    = (const float*)d_in[26];

  char* ws = (char*)d_ws;
  size_t off = 0;
  auto alloc = [&](size_t bytes) -> void* {
    void* p = ws + off;
    off += bytes;
    off = (off + 255) & ~(size_t)255;
    return p;
  };
  unsigned short* xb  = (unsigned short*)alloc((size_t)ROWS * DDIM * 2);
  unsigned short* tb  = (unsigned short*)alloc((size_t)ROWS * DDIM * 2);
  unsigned short* wt6 = (unsigned short*)alloc((size_t)6 * 262144 * 2);
  float* pooled = (float*)alloc((size_t)BNUM * DDIM * 4);
  float* gctx   = (float*)alloc((size_t)32 * DDIM * 4);
  float* tmpA   = (float*)alloc((size_t)NEWBN * 1536 * 4);
  float* tmpB   = (float*)alloc((size_t)NEWBN * 1536 * 4);
  float* allf   = (float*)alloc((size_t)NEWBN * DDIM * 4);
  float* tvec   = (float*)alloc((size_t)NEWBN * 3 * 4);
  float* rvec   = (float*)alloc((size_t)NEWBN * 9 * 4);

  // feat -> bf16
  int n4 = ROWS * DDIM / 4;
  k_cvt_bf16<<<(n4 + 255) / 256, 256, 0, stream>>>(feat, xb, n4);
  // weights -> transposed bf16
  int tw = 6 * 262144;
  k_wt_bf16<<<(tw + 255) / 256, 256, 0, stream>>>(W_res, wt6, tw);

  // two ResConvBlocks, 3 layers each; layer 3 fuses residual add into xb
  const int gblk = ROWS / 32;   // 4107
  for (int b = 0; b < 2; ++b) {
    int l0 = b * 3;
    k_gemm_wmma<<<gblk, 256, 0, stream>>>(xb, wt6 + (size_t)(l0 + 0) * 262144,
                                          b_res + (l0 + 0) * DDIM, tb, 0);
    k_gemm_wmma<<<gblk, 256, 0, stream>>>(tb, wt6 + (size_t)(l0 + 1) * 262144,
                                          b_res + (l0 + 1) * DDIM, tb, 0);
    k_gemm_wmma<<<gblk, 256, 0, stream>>>(tb, wt6 + (size_t)(l0 + 2) * 262144,
                                          b_res + (l0 + 2) * DDIM, xb, 1);
  }

  k_pool<<<BNUM, DDIM, 0, stream>>>(xb, pooled);

  // cur = Lin2(relu(Lin1(pooled)))  -> allf[0:96]
  k_sgemm<<<BNUM, 256, 0, stream>>>(pooled, W_cur1, b_cur1, tmpA, BNUM, 512, 512, 1);
  k_sgemm<<<BNUM, 256, 0, stream>>>(tmpA, W_cur2, b_cur2, allf, BNUM, 512, 512, 0);
  // gctx / temporal prediction / future
  k_gctx<<<(32 * DDIM + 255) / 256, 256, 0, stream>>>(pooled, gctx);
  k_sgemm<<<32, 256, 0, stream>>>(gctx, W_tp1, b_tp1, tmpA, 32, 512, 512, 1);
  k_sgemm<<<32, 256, 0, stream>>>(tmpA, W_tp2, b_tp2, tmpB, 32, 1536, 512, 0);  // tf -> [96,512]
  k_sgemm<<<BNUM, 256, 0, stream>>>(tmpB, W_fut1, b_fut1, tmpA, BNUM, 512, 512, 1);
  k_sgemm<<<BNUM, 256, 0, stream>>>(tmpA, W_fut2, b_fut2, allf + (size_t)BNUM * DDIM,
                                    BNUM, 512, 512, 0);
  // more_mlps on concat
  k_sgemm<<<NEWBN, 256, 0, stream>>>(allf, W_m1, b_m1, tmpA, NEWBN, 512, 512, 1);
  k_sgemm<<<NEWBN, 256, 0, stream>>>(tmpA, W_m2, b_m2, tmpB, NEWBN, 512, 512, 1);
  // pose heads
  k_sgemm<<<NEWBN, 64, 0, stream>>>(tmpB, W_t, b_t, tvec, NEWBN, 3, 512, 0);
  k_sgemm<<<NEWBN, 64, 0, stream>>>(tmpB, W_r, b_r, rvec, NEWBN, 9, 512, 0);
  // SVD orthogonalize + 4x4 assembly
  k_pose<<<(NEWBN + 63) / 64, 64, 0, stream>>>(tvec, rvec, (float*)d_out, NEWBN);
}
